// PreampStage_28810640621780
// MI455X (gfx1250) — compile-verified, matrix operand order checked by reading, not run on hardware
//
#include <hip/hip_runtime.h>
#include <stdint.h>

typedef __attribute__((ext_vector_type(16))) _Float16 v16h;
typedef __attribute__((ext_vector_type(8)))  float    v8f;
typedef __attribute__((ext_vector_type(4))) uint32_t u32x4;
typedef __attribute__((ext_vector_type(8)))  int32_t  i32x8;
typedef __attribute__((ext_vector_type(4)))  int32_t  i32x4;

#define BATCH 64
#define SEQ   65536
#define TILE  64
#define LSTR  65            // padded LDS row stride (floats) -> conflict-free banks
#define NTILE (SEQ / TILE)

// Hardware reciprocal (v_rcp_f32): 1 TRANS op instead of ~10-op IEEE div expansion.
__device__ __forceinline__ float frcp(float x) { return __builtin_amdgcn_rcpf(x); }
__device__ __forceinline__ float sigf(float x) { return frcp(1.0f + __expf(-x)); }
__device__ __forceinline__ float tanhfast(float x) {
  return 1.0f - 2.0f * frcp(__expf(2.0f * x) + 1.0f);
}

#define HAVE_TDM __has_builtin(__builtin_amdgcn_tensor_load_to_lds)

#if HAVE_TDM
// Issue one async TDM load: 32 rows x TILE cols of f32 from (gsrc, row stride SEQ)
// into LDS at byte address ldsByte, padded +1 dword per 64 dwords (-> stride 65).
__device__ __forceinline__ void tdm_issue(const float* gsrc, uint32_t ldsByte) {
  uint64_t ga = (uint64_t)(uintptr_t)gsrc;
  u32x4 g0;
  g0[0] = 1u;                                 // count=1 (valid), no gather
  g0[1] = ldsByte;                            // lds_addr
  g0[2] = (uint32_t)ga;                       // global_addr[31:0]
  g0[3] = (uint32_t)(ga >> 32) | (2u << 30);  // global_addr[56:32] | type=2
  i32x8 g1;
  g1[0] = (2 << 16) | (1 << 20) | (5 << 22);  // data_size=4B, pad_enable, interval=64dw, amount=1dw
  g1[1] = 0;                                  // atomic_barrier=0, tensor_dim0[15:0]=0 (65536)
  g1[2] = (SEQ >> 16) | (32 << 16);           // tensor_dim0[31:16]=1 ; tensor_dim1[15:0]=32
  g1[3] = (TILE << 16);                       // tensor_dim1 hi=0 ; tile_dim0=TILE
  g1[4] = 32;                                 // tile_dim1=32 rows ; tile_dim2=0
  g1[5] = SEQ;                                // tensor_dim0_stride low32
  g1[6] = 0;                                  // stride0 hi16 / stride1 lo16
  g1[7] = 0;
  i32x4 gz = {};
#if __has_include(<hip/amd_detail/amd_gfx1250_TDM.h>)
  i32x8 g4 = {};
  __builtin_amdgcn_tensor_load_to_lds(g0, g1, gz, gz, g4, 0);
#else
  __builtin_amdgcn_tensor_load_to_lds(g0, g1, gz, gz, 0);
#endif
}
#endif

__global__ void __launch_bounds__(64)
preamp_kernel(const float* __restrict__ x, const float* __restrict__ knobs,
              const float* __restrict__ pre_c, const float* __restrict__ post_c,
              const float* __restrict__ w_ih, const float* __restrict__ w_hh,
              const float* __restrict__ b_ih, const float* __restrict__ b_hh,
              const float* __restrict__ w_out, const float* __restrict__ b_out,
              const float* __restrict__ kw1, const float* __restrict__ kb1,
              const float* __restrict__ kw2, const float* __restrict__ kb2,
              float* __restrict__ out) {
  __shared__ float xs[2][BATCH * LSTR];   // double-buffered input tiles (padded)
  __shared__ float ys[BATCH * LSTR];      // output staging tile (padded)
  __shared__ float hb[BATCH][16];         // knob MLP hidden

  const int tid  = threadIdx.x;           // 0..63, one batch row per thread
  const int lane = tid & 31;
  const int wave = tid >> 5;

  // ---------- Knob MLP layer 1: h = tanh(knobs @ w1^T + b1) via WMMA ----------
  {
    float b1v = kb1[lane & 15];
    v8f cc;
#pragma unroll
    for (int i = 0; i < 8; ++i) cc[i] = b1v;        // C[m,n] = b1[n]
    v16h bm = {};                                    // B[0,n] = w1[n], rest 0
    float wv = kw1[lane & 15];
    bm[0] = (lane < 16) ? (_Float16)wv : (_Float16)0.0f;
#pragma unroll
    for (int ch = 0; ch < 2; ++ch) {                 // two 16-row chunks per wave
      v16h am = {};                                  // A[m,0] = knob[m], rest 0
      float kv = knobs[wave * 32 + ch * 16 + (lane & 15)];
      am[0] = (lane < 16) ? (_Float16)kv : (_Float16)0.0f;
      v8f d = __builtin_amdgcn_wmma_f32_16x16x32_f16(false, am, false, bm,
                                                     (short)0, cc, false, false);
      int mb = wave * 32 + ch * 16 + ((lane >> 4) << 3);   // D row base for this lane
      int n = lane & 15;                                   // D column
#pragma unroll
      for (int i = 0; i < 8; ++i) hb[mb + i][n] = tanhfast(d[i]);
    }
  }
  __syncthreads();

  // ---------- Knob MLP layer 2 (scalar per row) ----------
  float p0 = kb2[0], p1 = kb2[1];
#pragma unroll
  for (int j = 0; j < 16; ++j) {
    float hv = hb[tid][j];
    p0 = fmaf(hv, kw2[j], p0);
    p1 = fmaf(hv, kw2[16 + j], p1);
  }
  p0 = sigf(p0); p1 = sigf(p1);
  const float gain = __expf(p0 * 4.0f - 2.0f);
  const float bias = p1 * 0.1f;

  // ---------- uniform coefficients ----------
  const float A0 = pre_c[0], A1 = pre_c[1], A2 = pre_c[2], A3 = pre_c[3], A4 = pre_c[4];
  const float B0 = pre_c[5], B1 = pre_c[6], B2 = pre_c[7], B3 = pre_c[8], B4 = pre_c[9];
  const float C0 = post_c[0], C1 = post_c[1], C2 = post_c[2], C3 = post_c[3], C4 = post_c[4];
  const float D0 = post_c[5], D1 = post_c[6], D2 = post_c[7], D3 = post_c[8], D4 = post_c[9];
  const float wir = w_ih[0], wiz = w_ih[1], win = w_ih[2];
  const float whr = w_hh[0], whz = w_hh[1], whn = w_hh[2];
  const float bir = b_ih[0], biz = b_ih[1], bin_ = b_ih[2];
  const float bhr = b_hh[0], bhz = b_hh[1], bhn = b_hh[2];
  const float wo = w_out[0], bo = b_out[0];

  // ---------- recurrence state (registers) ----------
  float s1x1 = 0, s1x2 = 0, s1y1 = 0, s1y2 = 0;   // pre biquad 1
  float s2x1 = 0, s2x2 = 0, s2y1 = 0, s2y2 = 0;   // pre biquad 2
  float h = 0.0f;                                  // GRU hidden
  float s3x1 = 0, s3x2 = 0, s3y1 = 0, s3y2 = 0;   // post biquad 1
  float s4x1 = 0, s4x2 = 0, s4y1 = 0, s4y2 = 0;   // post biquad 2

#if HAVE_TDM
  const float* grow = x + (size_t)(wave * 32) * SEQ;
  const uint32_t ldsRowBase = (uint32_t)(uintptr_t)&xs[0][wave * 32 * LSTR];
  const uint32_t ldsBufStep = (uint32_t)(BATCH * LSTR * sizeof(float));
  tdm_issue(grow, ldsRowBase);                     // prologue: tile 0 -> buf 0
#endif

  for (int k = 0; k < NTILE; ++k) {
    const int t0 = k * TILE;
#if HAVE_TDM
    if (k + 1 < NTILE) {                           // prefetch next tile, wait on current
      tdm_issue(grow + (size_t)(k + 1) * TILE,
                ldsRowBase + (uint32_t)((k + 1) & 1) * ldsBufStep);
      __builtin_amdgcn_s_wait_tensorcnt(1);
    } else {
      __builtin_amdgcn_s_wait_tensorcnt(0);
    }
    __syncthreads();                               // ys reuse fence + tile ready
#else
    __syncthreads();
    for (int rr = 0; rr < BATCH; ++rr)             // coalesced fallback load
      xs[k & 1][rr * LSTR + tid] = x[(size_t)rr * SEQ + t0 + tid];
    __syncthreads();
#endif
    const float* xb = &xs[k & 1][tid * LSTR];
    float* yb = &ys[tid * LSTR];
#pragma unroll 4
    for (int t = 0; t < TILE; ++t) {
      float xv = xb[t];
      // pre biquad cascade (DF-I)
      float y1 = A0 * xv + A1 * s1x1 + A2 * s1x2 - A3 * s1y1 - A4 * s1y2;
      s1x2 = s1x1; s1x1 = xv; s1y2 = s1y1; s1y1 = y1;
      float y2 = B0 * y1 + B1 * s2x1 + B2 * s2x2 - B3 * s2y1 - B4 * s2y2;
      s2x2 = s2x1; s2x1 = y1; s2y2 = s2y1; s2y1 = y2;
      // gain/bias
      float u = fmaf(y2, gain, bias);
      // GRU (H=1), torch gate order r,z,n
      float gr = sigf(fmaf(u, wir, bir) + fmaf(h, whr, bhr));
      float gz = sigf(fmaf(u, wiz, biz) + fmaf(h, whz, bhz));
      float gn = tanhfast(fmaf(u, win, bin_) + gr * fmaf(h, whn, bhn));
      h = fmaf(1.0f - gz, gn, gz * h);
      float v = fmaf(h, wo, bo);
      // post biquad cascade
      float y3 = C0 * v + C1 * s3x1 + C2 * s3x2 - C3 * s3y1 - C4 * s3y2;
      s3x2 = s3x1; s3x1 = v; s3y2 = s3y1; s3y1 = y3;
      float y4 = D0 * y3 + D1 * s4x1 + D2 * s4x2 - D3 * s4y1 - D4 * s4y2;
      s4x2 = s4x1; s4x1 = y3; s4y2 = s4y1; s4y1 = y4;
      yb[t] = y4;
    }
    __syncthreads();
    // coalesced cooperative store of the 64x64 output tile
    for (int rr = 0; rr < BATCH; ++rr)
      out[(size_t)rr * SEQ + t0 + tid] = ys[rr * LSTR + tid];
  }
}

extern "C" void kernel_launch(void* const* d_in, const int* in_sizes, int n_in,
                              void* d_out, int out_size, void* d_ws, size_t ws_size,
                              hipStream_t stream) {
  (void)in_sizes; (void)n_in; (void)out_size; (void)d_ws; (void)ws_size;
  preamp_kernel<<<dim3(1), dim3(64), 0, stream>>>(
      (const float*)d_in[0],  (const float*)d_in[1],  (const float*)d_in[2],
      (const float*)d_in[3],  (const float*)d_in[4],  (const float*)d_in[5],
      (const float*)d_in[6],  (const float*)d_in[7],  (const float*)d_in[8],
      (const float*)d_in[9],  (const float*)d_in[10], (const float*)d_in[11],
      (const float*)d_in[12], (const float*)d_in[13], (float*)d_out);
}
